// CLIPAttention_84061099917818
// MI455X (gfx1250) — matrix operand order repaired and checked by reading, not compiled, not run on hardware
//
#include <hip/hip_runtime.h>
#include <hip/hip_bf16.h>

// CLIP attention for MI455X (gfx1250, wave32, WMMA).
// B=8, N=1024, C=1024, H=16, Dh=64.
//
// cvt(f32->f16) -> QKV WMMA GEMM -> fused flash attention (scores stay
// on-chip) -> out-proj WMMA GEMM.  All LDS staging is double-buffered via
// GLOBAL_LOAD_ASYNC_TO_LDS_B128 (ASYNCcnt) when available.

typedef __attribute__((ext_vector_type(16))) _Float16 v16h;
typedef __attribute__((ext_vector_type(8)))  float    v8f;
typedef __attribute__((ext_vector_type(4)))  int      v4i;

#define ASG __attribute__((address_space(1)))
#define ASL __attribute__((address_space(3)))

#if defined(__AMDGCN__) && \
    __has_builtin(__builtin_amdgcn_global_load_async_to_lds_b128) && \
    __has_builtin(__builtin_amdgcn_s_wait_asynccnt)
#define HAVE_ASYNC_LDS 1
#else
#define HAVE_ASYNC_LDS 0
#endif

#if HAVE_ASYNC_LDS
#define WAIT_ASYNC(n) __builtin_amdgcn_s_wait_asynccnt(n)
#else
#define WAIT_ASYNC(n) do { } while (0)
#endif

// ---------------------------------------------------------------- WMMA wrap
__device__ __forceinline__ v8f wmma_f32(v16h a, v16h b, v8f c) {
  return __builtin_amdgcn_wmma_f32_16x16x32_f16(
      /*neg_a=*/false, a, /*neg_b=*/false, b,
      /*c_mod=*/(short)0, c, /*reuse_a=*/false, /*reuse_b=*/false);
}

// ------------------------------------------------- fragment loaders (wave32)
// A-matrix 16x32 f16 (ISA 7.12.2): lane l holds row m=l&15; lanes 0-15 cover
// K={0..7,16..23}, lanes 16-31 cover K={8..15,24..31}.
__device__ __forceinline__ v16h frag_a(const _Float16* base, int ld, int lane) {
  const int m  = lane & 15;
  const int kb = (lane >> 4) << 3;
  v16h a;
#pragma unroll
  for (int h = 0; h < 16; ++h) {
    const int j2 = h >> 1;
    const int k  = kb + ((j2 & 3) << 1) + ((j2 & 4) << 2) + (h & 1);
    a[h] = base[m * ld + k];
  }
  return a;
}

// B-matrix 32x16 f16, source stored [n][k]: lane column n=l&15;
// lanes 0-15: K={0..7,16..23}, lanes 16-31: K={8..15,24..31} (matches A).
__device__ __forceinline__ v16h frag_b_nk(const _Float16* base, int ld, int lane) {
  const int n  = lane & 15;
  const int kb = (lane >> 4) << 3;
  v16h b;
#pragma unroll
  for (int h = 0; h < 16; ++h) b[h] = base[n * ld + kb + h + (h & 8)];
  return b;
}

// B-matrix 32x16 f16, source stored [k][n] (natural layout).
__device__ __forceinline__ v16h frag_b_kn(const _Float16* base, int ld, int lane) {
  const int n  = lane & 15;
  const int kb = (lane >> 4) << 3;
  v16h b;
#pragma unroll
  for (int h = 0; h < 16; ++h) b[h] = base[(kb + h + (h & 8)) * ld + n];
  return b;
}

// --------------------------------------------- global->LDS tile staging
// 16-byte chunks; LLD*2 must be a multiple of 16 (40,72 ok). Compile-time
// trip count (no exec divergence). Async: memory->LDS DMA, ASYNCcnt-tracked.
template <int ROWS, int COLS, int LLD, int NTHR>
__device__ __forceinline__ void stage_tile(_Float16* __restrict__ dst,
                                           const _Float16* __restrict__ src,
                                           int gld, int tid) {
  static_assert((ROWS * COLS / 8) % NTHR == 0, "chunk count");
  constexpr int CPT = (ROWS * COLS / 8) / NTHR;   // b128 chunks per thread
#if HAVE_ASYNC_LDS
#pragma unroll
  for (int i = 0; i < CPT; ++i) {
    const int c   = tid + i * NTHR;
    const int r   = c / (COLS / 8);
    const int col = (c % (COLS / 8)) * 8;
    __builtin_amdgcn_global_load_async_to_lds_b128(
        (ASG v4i*)(src + (size_t)r * gld + col),
        (ASL v4i*)(dst + r * LLD + col), 0, 0);
  }
#else
  uint4 tmp[CPT];
#pragma unroll
  for (int i = 0; i < CPT; ++i) {
    const int c   = tid + i * NTHR;
    const int r   = c / (COLS / 8);
    const int col = (c % (COLS / 8)) * 8;
    tmp[i] = *(const uint4*)(src + (size_t)r * gld + col);
  }
#pragma unroll
  for (int i = 0; i < CPT; ++i) {
    const int c   = tid + i * NTHR;
    const int r   = c / (COLS / 8);
    const int col = (c % (COLS / 8)) * 8;
    *(uint4*)(dst + r * LLD + col) = tmp[i];
  }
#endif
}

// ------------------------------------------------------------ f32 -> f16
__global__ void cvt_f32_f16_kernel(const float* __restrict__ src,
                                   _Float16* __restrict__ dst, int n) {
  int i = blockIdx.x * blockDim.x + threadIdx.x;
  if (i < n) dst[i] = (_Float16)src[i];
}

// ------------------------------------------------------------- QKV GEMM
// QKV[m,d] = sum_k X[m,k]*W[d,k] + bias[d]; M=8192, Nd=3072, K=1024.
// Epilogue scatters into head-major f16 Q/K/V [B,H,N,64]; Q gets *0.125.
__device__ __forceinline__ void store_qkv_tile(const v8f& c, int mbase, int nbase,
                                               int lane, const float* __restrict__ bias,
                                               _Float16* __restrict__ Q,
                                               _Float16* __restrict__ Kh,
                                               _Float16* __restrict__ Vh) {
  const int n    = nbase + (lane & 15);
  const int mofs = (lane >> 4) << 3;
  const float bi = bias[n];
  const int sec = n >> 10;            // 0:Q 1:K 2:V
  const int dd  = n & 1023;
  const int h   = dd >> 6;
  const int dh  = dd & 63;
  _Float16* dst = (sec == 0) ? Q : ((sec == 1) ? Kh : Vh);
  const float scale = (sec == 0) ? 0.125f : 1.0f;   // 1/sqrt(64) folded into Q
#pragma unroll
  for (int r = 0; r < 8; ++r) {
    const int m  = mbase + mofs + r;
    const int b  = m >> 10;
    const int nn = m & 1023;
    dst[(size_t)(((b * 16 + h) * 1024 + nn)) * 64 + dh] =
        (_Float16)((c[r] + bi) * scale);
  }
}

__global__ __launch_bounds__(256) void qkv_gemm_kernel(
    const _Float16* __restrict__ X, const _Float16* __restrict__ W,
    const float* __restrict__ bias, _Float16* __restrict__ Q,
    _Float16* __restrict__ Kh, _Float16* __restrict__ Vh) {
  constexpr int BM = 128, BN = 64, BK = 32, LLD = BK + 8;
  constexpr int ASZ = BM * LLD, BSZ = BN * LLD;
  constexpr int NT  = 1024 / BK;
  __shared__ _Float16 As[2 * ASZ];
  __shared__ _Float16 Bs[2 * BSZ];
  const int tid = threadIdx.x, lane = tid & 31, wave = tid >> 5;
  const int wm = wave >> 1, wn = wave & 1;        // waves tiled 4x2
  const int m0 = blockIdx.y * BM, n0 = blockIdx.x * BN;
  const _Float16* Xb = X + (size_t)m0 * 1024;
  const _Float16* Wb = W + (size_t)n0 * 1024;

  v8f c00 = {}, c01 = {}, c10 = {}, c11 = {};

  // prologue: stage tile 0 into buffer 0 (3 async ops / thread)
  stage_tile<BM, BK, LLD, 256>(As, Xb, 1024, tid);
  stage_tile<BN, BK, LLD, 256>(Bs, Wb, 1024, tid);

  for (int t = 0; t < NT; ++t) {
    const int cur = (t & 1), nxt = cur ^ 1;
    if (t + 1 < NT) {   // stage next tile into the other buffer, then wait
      stage_tile<BM, BK, LLD, 256>(As + nxt * ASZ, Xb + (t + 1) * BK, 1024, tid);
      stage_tile<BN, BK, LLD, 256>(Bs + nxt * BSZ, Wb + (t + 1) * BK, 1024, tid);
      WAIT_ASYNC(3);    // in-order async loads: tile t is resident
    } else {
      WAIT_ASYNC(0);
    }
    __syncthreads();    // tile t visible to all waves
    const _Float16* as = As + cur * ASZ;
    const _Float16* bs = Bs + cur * BSZ;
    v16h a0 = frag_a(as + (wm * 32) * LLD, LLD, lane);
    v16h a1 = frag_a(as + (wm * 32 + 16) * LLD, LLD, lane);
    v16h b0 = frag_b_nk(bs + (wn * 32) * LLD, LLD, lane);
    v16h b1 = frag_b_nk(bs + (wn * 32 + 16) * LLD, LLD, lane);
    c00 = wmma_f32(a0, b0, c00);
    c01 = wmma_f32(a0, b1, c01);
    c10 = wmma_f32(a1, b0, c10);
    c11 = wmma_f32(a1, b1, c11);
    __syncthreads();    // all waves done reading before buffer reuse
  }
  store_qkv_tile(c00, m0 + wm * 32,      n0 + wn * 32,      lane, bias, Q, Kh, Vh);
  store_qkv_tile(c01, m0 + wm * 32,      n0 + wn * 32 + 16, lane, bias, Q, Kh, Vh);
  store_qkv_tile(c10, m0 + wm * 32 + 16, n0 + wn * 32,      lane, bias, Q, Kh, Vh);
  store_qkv_tile(c11, m0 + wm * 32 + 16, n0 + wn * 32 + 16, lane, bias, Q, Kh, Vh);
}

// --------------------------------------------------- fused flash attention
// grid = (B*H, N/64); 128 threads = 4 waves, each wave owns a 16-query tile;
// 32 keys/values per step, double-buffered in LDS.
__global__ __launch_bounds__(128) void attn_kernel(
    const _Float16* __restrict__ Q, const _Float16* __restrict__ K,
    const _Float16* __restrict__ V, _Float16* __restrict__ AO) {
  constexpr int LKV = 72;            // 64 + 8 pad; 144B rows (16B aligned)
  constexpr int LP  = 40;            // 32 + 8 pad
  constexpr int KSZ = 32 * LKV;
  __shared__ _Float16 Ks[2 * KSZ];
  __shared__ _Float16 Vs[2 * KSZ];
  __shared__ _Float16 Ps[4][16 * LP];

  const int bh  = blockIdx.x;        // b*16 + h
  const int tid = threadIdx.x, lane = tid & 31, wave = tid >> 5;
  const int q0  = blockIdx.y * 64 + wave * 16;
  const _Float16* Qp = Q + (size_t)bh * 1024 * 64;
  const _Float16* Kp = K + (size_t)bh * 1024 * 64;
  const _Float16* Vp = V + (size_t)bh * 1024 * 64;

  // Q tile as two A fragments (Dh=64 -> K 0..31 / 32..63), loaded once.
  const int mrow = lane & 15;
  const int kb   = (lane >> 4) << 3;
  v16h qa0, qa1;
#pragma unroll
  for (int h = 0; h < 16; ++h) {
    const int j2 = h >> 1;
    const int k  = kb + ((j2 & 3) << 1) + ((j2 & 4) << 2) + (h & 1);
    qa0[h] = Qp[(size_t)(q0 + mrow) * 64 + k];
    qa1[h] = Qp[(size_t)(q0 + mrow) * 64 + k + 32];
  }

  v8f o0 = {}, o1 = {}, o2 = {}, o3 = {};        // 16x64 f32 accumulators
  float rmax[8], rsum[8];
#pragma unroll
  for (int r = 0; r < 8; ++r) { rmax[r] = -1e30f; rsum[r] = 0.0f; }

  const int mofs = (lane >> 4) << 3;  // C-layout: lanes 16-31 hold rows r+8
  const int ncol = lane & 15;

  constexpr int NT = 1024 / 32;
  stage_tile<32, 64, LKV, 128>(Ks, Kp, 64, tid);       // 2 ops
  stage_tile<32, 64, LKV, 128>(Vs, Vp, 64, tid);       // 2 ops

  for (int t = 0; t < NT; ++t) {
    const int cur = (t & 1), nxt = cur ^ 1;
    if (t + 1 < NT) {
      stage_tile<32, 64, LKV, 128>(Ks + nxt * KSZ, Kp + (size_t)(t + 1) * 32 * 64, 64, tid);
      stage_tile<32, 64, LKV, 128>(Vs + nxt * KSZ, Vp + (size_t)(t + 1) * 32 * 64, 64, tid);
      WAIT_ASYNC(4);
    } else {
      WAIT_ASYNC(0);
    }
    __syncthreads();
    const _Float16* ks = Ks + cur * KSZ;
    const _Float16* vs = Vs + cur * KSZ;

    // S0: keys t*32..+15, S1: +16..+31   (B = K-tile^T in [n][k] form)
    v8f s0 = {}, s1 = {};
    s0 = wmma_f32(qa0, frag_b_nk(ks, LKV, lane), s0);
    s0 = wmma_f32(qa1, frag_b_nk(ks + 32, LKV, lane), s0);
    s1 = wmma_f32(qa0, frag_b_nk(ks + 16 * LKV, LKV, lane), s1);
    s1 = wmma_f32(qa1, frag_b_nk(ks + 16 * LKV + 32, LKV, lane), s1);

    // Online softmax: row reductions across the 16 lanes sharing a row.
    float mnew[8], alpha[8];
#pragma unroll
    for (int r = 0; r < 8; ++r) {
      float cmx = fmaxf(s0[r], s1[r]);
      cmx = fmaxf(cmx, __shfl_xor(cmx, 1));
      cmx = fmaxf(cmx, __shfl_xor(cmx, 2));
      cmx = fmaxf(cmx, __shfl_xor(cmx, 4));
      cmx = fmaxf(cmx, __shfl_xor(cmx, 8));
      mnew[r]  = fmaxf(rmax[r], cmx);
      alpha[r] = __expf(rmax[r] - mnew[r]);
      rmax[r]  = mnew[r];
    }
#pragma unroll
    for (int r = 0; r < 8; ++r) {
      const float p0 = __expf(s0[r] - mnew[r]);
      const float p1 = __expf(s1[r] - mnew[r]);
      s0[r] = p0; s1[r] = p1;
      float sm = p0 + p1;
      sm += __shfl_xor(sm, 1);
      sm += __shfl_xor(sm, 2);
      sm += __shfl_xor(sm, 4);
      sm += __shfl_xor(sm, 8);
      rsum[r] = rsum[r] * alpha[r] + sm;
      o0[r] *= alpha[r]; o1[r] *= alpha[r]; o2[r] *= alpha[r]; o3[r] *= alpha[r];
    }

    // Re-layout P (C-layout f32) -> A-fragment f16 via per-wave LDS scratch.
    _Float16* ps = Ps[wave];
#pragma unroll
    for (int r = 0; r < 8; ++r) {
      ps[(mofs + r) * LP + ncol]      = (_Float16)s0[r];
      ps[(mofs + r) * LP + ncol + 16] = (_Float16)s1[r];
    }
    v16h pa = frag_a(ps, LP, lane);   // same-wave DS RAW: in-order LDS

    // O += P(16x32) * V(32x64), four 16-column chunks of V ([k][n] form).
    o0 = wmma_f32(pa, frag_b_kn(vs +  0, LKV, lane), o0);
    o1 = wmma_f32(pa, frag_b_kn(vs + 16, LKV, lane), o1);
    o2 = wmma_f32(pa, frag_b_kn(vs + 32, LKV, lane), o2);
    o3 = wmma_f32(pa, frag_b_kn(vs + 48, LKV, lane), o3);
    __syncthreads();
  }

  // Normalize and store f16 attention output in [B, N, C] (C = h*64 + d).
  const int b = bh >> 4, h = bh & 15;
  float inv[8];
#pragma unroll
  for (int r = 0; r < 8; ++r) inv[r] = 1.0f / rsum[r];
#pragma unroll
  for (int r = 0; r < 8; ++r) {
    const size_t q = q0 + mofs + r;
    _Float16* dst = AO + ((size_t)b * 1024 + q) * 1024 + h * 64;
    dst[ 0 + ncol] = (_Float16)(o0[r] * inv[r]);
    dst[16 + ncol] = (_Float16)(o1[r] * inv[r]);
    dst[32 + ncol] = (_Float16)(o2[r] * inv[r]);
    dst[48 + ncol] = (_Float16)(o3[r] * inv[r]);
  }
}

// ------------------------------------------------------------ out-proj GEMM
// out[m,n] = sum_k AO[m,k]*Wout[n,k] + bias[n]  (fp32 result)
__global__ __launch_bounds__(256) void outproj_gemm_kernel(
    const _Float16* __restrict__ A, const _Float16* __restrict__ W,
    const float* __restrict__ bias, float* __restrict__ out) {
  constexpr int BM = 128, BN = 64, BK = 32, LLD = BK + 8;
  constexpr int ASZ = BM * LLD, BSZ = BN * LLD;
  constexpr int NT  = 1024 / BK;
  __shared__ _Float16 As[2 * ASZ];
  __shared__ _Float16 Bs[2 * BSZ];
  const int tid = threadIdx.x, lane = tid & 31, wave = tid >> 5;
  const int wm = wave >> 1, wn = wave & 1;
  const int m0 = blockIdx.y * BM, n0 = blockIdx.x * BN;
  const _Float16* Ab = A + (size_t)m0 * 1024;
  const _Float16* Wb = W + (size_t)n0 * 1024;

  v8f c00 = {}, c01 = {}, c10 = {}, c11 = {};
  stage_tile<BM, BK, LLD, 256>(As, Ab, 1024, tid);
  stage_tile<BN, BK, LLD, 256>(Bs, Wb, 1024, tid);

  for (int t = 0; t < NT; ++t) {
    const int cur = (t & 1), nxt = cur ^ 1;
    if (t + 1 < NT) {
      stage_tile<BM, BK, LLD, 256>(As + nxt * ASZ, Ab + (t + 1) * BK, 1024, tid);
      stage_tile<BN, BK, LLD, 256>(Bs + nxt * BSZ, Wb + (t + 1) * BK, 1024, tid);
      WAIT_ASYNC(3);
    } else {
      WAIT_ASYNC(0);
    }
    __syncthreads();
    const _Float16* as = As + cur * ASZ;
    const _Float16* bs = Bs + cur * BSZ;
    v16h a0 = frag_a(as + (wm * 32) * LLD, LLD, lane);
    v16h a1 = frag_a(as + (wm * 32 + 16) * LLD, LLD, lane);
    v16h b0 = frag_b_nk(bs + (wn * 32) * LLD, LLD, lane);
    v16h b1 = frag_b_nk(bs + (wn * 32 + 16) * LLD, LLD, lane);
    c00 = wmma_f32(a0, b0, c00);
    c01 = wmma_f32(a0, b1, c01);
    c10 = wmma_f32(a1, b0, c10);
    c11 = wmma_f32(a1, b1, c11);
    __syncthreads();
  }

  const int mofs = (lane >> 4) << 3;
  const int nc   = lane & 15;
#pragma unroll
  for (int t = 0; t < 4; ++t) {
    const v8f& c = (t == 0) ? c00 : (t == 1) ? c01 : (t == 2) ? c10 : c11;
    const int mbase = m0 + wm * 32 + ((t >> 1) << 4);
    const int n     = n0 + wn * 32 + ((t & 1) << 4) + nc;
    const float bi  = bias[n];
#pragma unroll
    for (int r = 0; r < 8; ++r)
      out[(size_t)(mbase + mofs + r) * 1024 + n] = c[r] + bi;
  }
}

// ------------------------------------------------------------------ launch
extern "C" void kernel_launch(void* const* d_in, const int* in_sizes, int n_in,
                              void* d_out, int out_size, void* d_ws, size_t ws_size,
                              hipStream_t stream) {
  const float* x     = (const float*)d_in[0];   // [8,1024,1024]
  const float* w_in  = (const float*)d_in[1];   // [3072,1024]
  const float* b_in  = (const float*)d_in[2];   // [3072]
  const float* w_out = (const float*)d_in[3];   // [1024,1024]
  const float* b_out = (const float*)d_in[4];   // [1024]
  float* out = (float*)d_out;                   // [8,1024,1024] fp32

  char* ws = (char*)d_ws;                       // ~93 MB of slabs
  size_t ofs = 0;
  auto carve = [&](size_t bytes) -> char* {
    char* p = ws + ofs;
    ofs = (ofs + bytes + 255) & ~(size_t)255;
    return p;
  };
  const size_t NX  = (size_t)8192 * 1024;
  const size_t NWI = (size_t)3072 * 1024;
  const size_t NWO = (size_t)1024 * 1024;
  const size_t NQ  = (size_t)8 * 16 * 1024 * 64;
  _Float16* xf  = (_Float16*)carve(NX  * 2);
  _Float16* wif = (_Float16*)carve(NWI * 2);
  _Float16* wof = (_Float16*)carve(NWO * 2);
  _Float16* Qf  = (_Float16*)carve(NQ  * 2);
  _Float16* Kf  = (_Float16*)carve(NQ  * 2);
  _Float16* Vf  = (_Float16*)carve(NQ  * 2);
  _Float16* AOf = (_Float16*)carve(NX  * 2);
  (void)ws_size; (void)n_in; (void)out_size; (void)in_sizes;

  cvt_f32_f16_kernel<<<(int)((NX  + 255) / 256), 256, 0, stream>>>(x, xf, (int)NX);
  cvt_f32_f16_kernel<<<(int)((NWI + 255) / 256), 256, 0, stream>>>(w_in, wif, (int)NWI);
  cvt_f32_f16_kernel<<<(int)((NWO + 255) / 256), 256, 0, stream>>>(w_out, wof, (int)NWO);

  qkv_gemm_kernel<<<dim3(3072 / 64, 8192 / 128), 256, 0, stream>>>(
      xf, wif, b_in, Qf, Kf, Vf);

  attn_kernel<<<dim3(128, 16), 128, 0, stream>>>(Qf, Kf, Vf, AOf);

  outproj_gemm_kernel<<<dim3(1024 / 64, 8192 / 128), 256, 0, stream>>>(
      AOf, wof, b_out, out);
}
